// CRFNew_58866821759207
// MI455X (gfx1250) — compile-verified, matrix operand order checked by reading, not run on hardware
//
#include <hip/hip_runtime.h>

typedef __attribute__((ext_vector_type(16))) _Float16 v16h;
typedef __attribute__((ext_vector_type(8)))  _Float16 v8h;
typedef __attribute__((ext_vector_type(8)))  float    v8f;

#define TB  32      // batch
#define TT  2048    // time steps
#define KK  256     // labels
#define BPB 16      // batches per block (one 16-row M tile)

#define LOG2E 1.44269504088896340736f
#define LN2   0.69314718055994530942f

// Raw HW transcendentals (v_log_f32 / v_exp_f32, base-2). Safe here:
// q in [0.95, 269] (never denormal/nonpositive), exp args are <= 0.
__device__ __forceinline__ float fast_log(float x) {
  return __builtin_amdgcn_logf(x) * LN2;
}
__device__ __forceinline__ float fast_exp(float x) {
  return __builtin_amdgcn_exp2f(x * LOG2E);
}

// ---------------------------------------------------------------------------
// Forward recursion: state[j] = logsumexp_i(state[i]+trans[i,j]) + x_t[j]
// reformulated as Q = P * Et  (P = exp(state - rowmax) f16, Et = exp(trans) f16)
// then state[j] = log(Q[j]) + rowmax + x_t[j].
// Block = 16 waves (512 thr). Wave w owns output tile (16 batches x cols 16w..16w+15).
// Et tiles are register-resident (8 x v16h per wave). p is double-buffered in LDS.
// ---------------------------------------------------------------------------
__launch_bounds__(512, 1)
__global__ void crf_forward_kernel(const float* __restrict__ yp,
                                   const float* __restrict__ trans,
                                   const float* __restrict__ path,
                                   float* __restrict__ out)
{
  __shared__ __align__(16) _Float16 p_half[2][BPB][KK];  // 16 KB
  __shared__ __align__(16) float    s_buf[2][BPB][KK];   // 32 KB
  __shared__ __align__(32) float    rowmax[2][BPB];

  const int tid     = threadIdx.x;
  const int wave    = tid >> 5;     // 0..15 : N-tile index
  const int lane    = tid & 31;
  const int hi      = lane >> 4;    // half-wave selector in WMMA layout
  const int ln      = lane & 15;
  const int b0      = blockIdx.x * BPB;
  const int colbase = wave * 16;

  // ---- Et (B matrix) tiles into registers, WMMA 16-bit B 32x16 layout ----
  v16h Breg[8];
  #pragma unroll
  for (int kb = 0; kb < 8; ++kb) {
    union { v16h v; _Float16 h[16]; } u;
    #pragma unroll
    for (int h = 0; h < 16; ++h) {
      const int klocal = (h < 8) ? (hi * 8 + h) : (16 + hi * 8 + (h - 8));
      const int k = kb * 32 + klocal;
      u.h[h] = (_Float16)fast_exp(trans[k * KK + colbase + ln]);
    }
    Breg[kb] = u.v;
  }

  // ---- t = 0: state = y_pred[:,0,:]; wave w postprocesses row m = w ----
  {
    const int kbase = lane * 8;
    float v[8]; float mx = -3.0e38f;
    #pragma unroll
    for (int i = 0; i < 8; ++i) {
      v[i] = yp[(size_t)(b0 + wave) * (TT * KK) + kbase + i];
      mx = fmaxf(mx, v[i]);
    }
    #pragma unroll
    for (int off = 16; off > 0; off >>= 1) mx = fmaxf(mx, __shfl_xor(mx, off, 32));
    union { v8f v; float f[8]; } sf;
    union { v8h v; _Float16 h[8]; } ph;
    #pragma unroll
    for (int i = 0; i < 8; ++i) {
      sf.f[i] = v[i];
      ph.h[i] = (_Float16)fast_exp(v[i] - mx);
    }
    *(v8f*)&s_buf[0][wave][kbase]  = sf.v;
    *(v8h*)&p_half[0][wave][kbase] = ph.v;
    if (lane == 0) rowmax[0][wave] = mx;
  }
  __syncthreads();

  // ---- sequential recursion ----
  for (int t = 1; t < TT; ++t) {
    const int nb = t & 1, pb = nb ^ 1;

    // emission loads for this step (independent of p; issued early)
    float y[8];
    #pragma unroll
    for (int r = 0; r < 8; ++r) {
      const int m = r + 8 * hi;
      y[r] = yp[(size_t)(b0 + m) * (TT * KK) + (size_t)t * KK + colbase + ln];
    }

    // previous-step row offsets for this lane's 8 rows (vector LDS read)
    const v8f mxv = *(const v8f*)&rowmax[pb][8 * hi];

    // A tiles from LDS (16-bit A 16x32 layout), accumulate K=256 via 8 WMMAs
    v8f c = {};
    #pragma unroll
    for (int kb = 0; kb < 8; ++kb) {
      union { v16h v; v8h h2[2]; } a;
      a.h2[0] = *(const v8h*)&p_half[pb][ln][kb * 32 + hi * 8];
      a.h2[1] = *(const v8h*)&p_half[pb][ln][kb * 32 + 16 + hi * 8];
      c = __builtin_amdgcn_wmma_f32_16x16x32_f16(false, a.v, false, Breg[kb],
                                                 (short)0, c, false, false);
    }

    // new state: s = log(q) + rowmax_prev + x_t  (q in [0.95, 269], log safe)
    #pragma unroll
    for (int r = 0; r < 8; ++r) {
      const int m = r + 8 * hi;
      s_buf[nb][m][colbase + ln] = fast_log(c[r]) + mxv[r] + y[r];
    }
    __syncthreads();  // all s stored

    // postprocess: wave w owns row m = w -> p = exp(s - max), rowmax
    {
      const int kbase = lane * 8;
      float v[8]; float mx = -3.0e38f;
      #pragma unroll
      for (int i = 0; i < 8; ++i) {
        v[i] = s_buf[nb][wave][kbase + i];
        mx = fmaxf(mx, v[i]);
      }
      #pragma unroll
      for (int off = 16; off > 0; off >>= 1) mx = fmaxf(mx, __shfl_xor(mx, off, 32));
      union { v8h v; _Float16 h[8]; } ph;
      #pragma unroll
      for (int i = 0; i < 8; ++i) ph.h[i] = (_Float16)fast_exp(v[i] - mx);
      *(v8h*)&p_half[nb][wave][kbase] = ph.v;
      if (lane == 0) rowmax[nb][wave] = mx;
    }
    __syncthreads();  // new p / rowmax visible
  }

  // ---- finalize: log_norm = logsumexp(state_final), out = log_norm - path ----
  {
    const int nb = (TT - 1) & 1;
    const int kbase = lane * 8;
    float v[8]; float mx = -3.0e38f;
    #pragma unroll
    for (int i = 0; i < 8; ++i) {
      v[i] = s_buf[nb][wave][kbase + i];
      mx = fmaxf(mx, v[i]);
    }
    #pragma unroll
    for (int off = 16; off > 0; off >>= 1) mx = fmaxf(mx, __shfl_xor(mx, off, 32));
    float sum = 0.f;
    #pragma unroll
    for (int i = 0; i < 8; ++i) sum += fast_exp(v[i] - mx);
    #pragma unroll
    for (int off = 16; off > 0; off >>= 1) sum += __shfl_xor(sum, off, 32);
    if (lane == 0) out[b0 + wave] = (mx + fast_log(sum)) - path[b0 + wave];
  }
}

// ---------------------------------------------------------------------------
// path_score[b] = sum_t y_pred[b,t,labels[b,t]] + sum_t trans[l_t, l_{t+1}]
// ---------------------------------------------------------------------------
__global__ void crf_path_kernel(const float* __restrict__ yp,
                                const float* __restrict__ trans,
                                const int* __restrict__ labels,
                                float* __restrict__ path)
{
  const int b   = blockIdx.x;
  const int tid = threadIdx.x;
  float acc = 0.f;
  for (int t = tid; t < TT; t += 256) {
    const int l = labels[b * TT + t];
    acc += yp[(size_t)b * (TT * KK) + (size_t)t * KK + l];
    if (t + 1 < TT) {
      const int l2 = labels[b * TT + t + 1];
      acc += trans[l * KK + l2];
    }
  }
  __shared__ float red[256];
  red[tid] = acc;
  __syncthreads();
  for (int s = 128; s > 0; s >>= 1) {
    if (tid < s) red[tid] += red[tid + s];
    __syncthreads();
  }
  if (tid == 0) path[b] = red[0];
}

extern "C" void kernel_launch(void* const* d_in, const int* in_sizes, int n_in,
                              void* d_out, int out_size, void* d_ws, size_t ws_size,
                              hipStream_t stream) {
  const float* yp     = (const float*)d_in[0];   // (32, 2048, 256) f32
  const float* trans  = (const float*)d_in[1];   // (256, 256) f32
  const int*   labels = (const int*)d_in[2];     // (32, 2048) i32
  float* out  = (float*)d_out;                   // (32, 1) f32
  float* path = (float*)d_ws;                    // scratch: 32 floats

  crf_path_kernel<<<TB, 256, 0, stream>>>(yp, trans, labels, path);
  crf_forward_kernel<<<TB / BPB, 512, 0, stream>>>(yp, trans, path, out);
}
